// MultiQueryAttention_68350109548804
// MI455X (gfx1250) — matrix operand order repaired and checked by reading, not compiled, not run on hardware
//
#include <hip/hip_runtime.h>

// ---------------------------------------------------------------------------
// MQA layer for MI455X (gfx1250, wave32, WMMA bf16 16x16x32).
//   x:(2,2048,2048) f32, sin/cos:(2048,64), doc_ids:(2,2048) i32,
//   W_qkv:(2048,2304), W_out:(2048,2048)  ->  out:(2,2048,2048) f32
// Fragment-major LDS layouts: each lane's 16 bf16 frag elements are 32
// contiguous bytes -> fragment load = 2x ds_load_b128.  Pure-copy tiles are
// staged with global_load_async_to_lds_b128 (ASYNCcnt).
// ---------------------------------------------------------------------------

typedef unsigned short u16;
typedef __attribute__((ext_vector_type(16))) __bf16 bh16;
typedef __attribute__((ext_vector_type(8)))  float  f32x8;

#define D_MODEL   2048
#define HEAD_DIM  128
#define NUM_HEADS 16
#define SEQ_T     2048
#define QKV_N     2304

union FragU  { bh16 v; u16 u[16]; };
union FragQ  { bh16 v; uint4 q[2]; };

__device__ __forceinline__ u16 f2bf(float x) {
  unsigned u = __builtin_bit_cast(unsigned, x);
  u += 0x7FFFu + ((u >> 16) & 1u);      // round-to-nearest-even
  return (u16)(u >> 16);
}

__device__ __forceinline__ f32x8 zero8() {
  f32x8 z;
  #pragma unroll
  for (int i = 0; i < 8; ++i) z[i] = 0.0f;
  return z;
}

// Element index inside a lane's 16-elem A-fragment run for k in [0,32)
// (ISA 16-bit A layout); lane = (m&15) + 16*((k>>3)&1).
__device__ __forceinline__ int a_elem(int k) {
  return 2 * (((k >> 4) << 2) + ((k >> 1) & 3)) + (k & 1);
}
// B layout: lane = (n&15) + 16*((k>>3)&1); elem = a_elem(k).

// Load one 16x16x32-operand fragment from a frag-major 512-u16 block:
// block = [lane 0..31][elem 0..15], 32B per lane -> 2x ds_load_b128.
__device__ __forceinline__ bh16 load_frag(const u16* block) {
  int lane = threadIdx.x & 31;
  const uint4* p = (const uint4*)(block + lane * 16);
  FragQ f;
  f.q[0] = p[0];
  f.q[1] = p[1];
  return f.v;
}

// Gather an A fragment (16x32) from row-major u16 (used once for Q, global).
__device__ __forceinline__ bh16 gather_a_frag(const u16* t, int lda) {
  int lane = threadIdx.x & 31;
  int m = lane & 15, kh = lane >> 4;
  FragU f;
  #pragma unroll
  for (int v = 0; v < 8; ++v) {
    int k = ((v >> 2) << 4) + (kh << 3) + ((v & 3) << 1);
    f.u[2 * v]     = t[m * lda + k];
    f.u[2 * v + 1] = t[m * lda + k + 1];
  }
  return f.v;
}

#define WMMA_BF16(a, b, c) \
  __builtin_amdgcn_wmma_f32_16x16x32_bf16(false, (a), false, (b), (short)0, (c), false, false)

__device__ __forceinline__ void async_b128(unsigned lds_off, const void* gsrc) {
  unsigned long long ga = (unsigned long long)gsrc;
  asm volatile("global_load_async_to_lds_b128 %0, %1, off"
               :: "v"(lds_off), "v"(ga) : "memory");
}
__device__ __forceinline__ void wait_async0() {
  asm volatile("s_wait_asynccnt 0x0" ::: "memory");
}
__device__ __forceinline__ unsigned lds_addr(const void* p) {
  return (unsigned)(unsigned long long)p;   // low 32 bits = LDS byte offset
}

// ---------------------------------------------------------------------------
// Kernel 1: qkv = x @ W_qkv (bf16 WMMA) + RoPE epilogue.
// Tile 64x128, K-step 32. grid=(64,18): y<16 Q head, ==16 K head, ==17 V head.
// ---------------------------------------------------------------------------
__global__ __launch_bounds__(256) void qkv_rope_kernel(
    const float* __restrict__ x, const float* __restrict__ Wqkv,
    const float* __restrict__ sinp, const float* __restrict__ cosp,
    u16* __restrict__ qb, u16* __restrict__ kb, u16* __restrict__ vb) {
  __shared__ __align__(32) float smem[64 * 129];   // 33 KB; C tile in epilogue
  u16* Af = (u16*)smem;                            // [4 sub][32 lane][16] = 2048
  u16* Bf = Af + 2048;                             // [8 sub][32 lane][16] = 4096

  const int row0 = blockIdx.x * 64;
  const int n0   = blockIdx.y * 128;
  const int tid  = threadIdx.x;
  const int wave = tid >> 5, wm = wave & 3, wn = wave >> 2;
  const int lane = tid & 31, kh = lane >> 4, nn = lane & 15;

  f32x8 acc[4];
  #pragma unroll
  for (int t = 0; t < 4; ++t) acc[t] = zero8();

  for (int k0 = 0; k0 < D_MODEL; k0 += 32) {
    #pragma unroll
    for (int i = 0; i < 8; ++i) {                  // A: 64x32 f32 -> frag-major
      int e = tid + 256 * i;
      int r = e >> 5, c = e & 31;
      int dst = ((r >> 4) * 32 + (r & 15) + (((c >> 3) & 1) << 4)) * 16 + a_elem(c);
      Af[dst] = f2bf(x[(size_t)(row0 + r) * D_MODEL + k0 + c]);
    }
    #pragma unroll
    for (int i = 0; i < 16; ++i) {                 // B: 32x128 -> frag-major
      int e = tid + 256 * i;
      int r = e >> 7, c = e & 127;                 // r=k, c=n
      int dst = ((c >> 4) * 32 + (c & 15) + (((r >> 3) & 1) << 4)) * 16 + a_elem(r);
      Bf[dst] = f2bf(Wqkv[(size_t)(k0 + r) * QKV_N + n0 + c]);
    }
    __syncthreads();
    bh16 a = load_frag(Af + wm * 512);
    #pragma unroll
    for (int t = 0; t < 4; ++t) {
      bh16 b = load_frag(Bf + (wn * 4 + t) * 512);
      acc[t] = WMMA_BF16(a, b, acc[t]);
    }
    __syncthreads();
  }

  // Spill C tile to LDS f32, then RoPE / convert.
  float* Cs = smem;                                // 64 x 129
  #pragma unroll
  for (int t = 0; t < 4; ++t)
    #pragma unroll
    for (int i = 0; i < 8; ++i) {
      int m = i + 8 * kh;
      Cs[(16 * wm + m) * 129 + 64 * wn + 16 * t + nn] = acc[t][i];
    }
  __syncthreads();

  if (n0 < D_MODEL) {                              // Q head
    int h = n0 >> 7;
    #pragma unroll
    for (int i = 0; i < 16; ++i) {
      int e = tid + 256 * i;
      int r = e >> 6, c = e & 63;
      int rf = row0 + r, b = rf >> 11, tt = rf & (SEQ_T - 1);
      float s = sinp[tt * 64 + c], cz = cosp[tt * 64 + c];
      float x1 = Cs[r * 129 + c], x2 = Cs[r * 129 + c + 64];
      u16* dst = qb + (((size_t)(b * NUM_HEADS + h) * SEQ_T + tt) * HEAD_DIM);
      dst[c]      = f2bf(x1 * cz - x2 * s);
      dst[c + 64] = f2bf(x2 * cz + x1 * s);
    }
  } else if (n0 == D_MODEL) {                      // K head
    #pragma unroll
    for (int i = 0; i < 16; ++i) {
      int e = tid + 256 * i;
      int r = e >> 6, c = e & 63;
      int rf = row0 + r, tt = rf & (SEQ_T - 1);
      float s = sinp[tt * 64 + c], cz = cosp[tt * 64 + c];
      float x1 = Cs[r * 129 + c], x2 = Cs[r * 129 + c + 64];
      u16* dst = kb + (size_t)rf * HEAD_DIM;
      dst[c]      = f2bf(x1 * cz - x2 * s);
      dst[c + 64] = f2bf(x2 * cz + x1 * s);
    }
  } else {                                         // V head
    #pragma unroll
    for (int i = 0; i < 32; ++i) {
      int e = tid + 256 * i;
      int r = e >> 7, c = e & 127;
      vb[(size_t)(row0 + r) * HEAD_DIM + c] = f2bf(Cs[r * 129 + c]);
    }
  }
}

// ---------------------------------------------------------------------------
// Kernel 2: flash attention. grid (B*H=32, T/64=32), 256 threads.
// ---------------------------------------------------------------------------
__global__ __launch_bounds__(256) void attn_kernel(
    const u16* __restrict__ qb, const u16* __restrict__ kb,
    const u16* __restrict__ vb, const int* __restrict__ doc,
    u16* __restrict__ ao) {
  __shared__ __align__(32) u16 Ktf[8192];  // [4 dchunk][4 nsub][32][16] 16 KB
  __shared__ __align__(32) u16 Vf[8192];   // [2 kc][8 nsub][32][16]    16 KB
  __shared__ __align__(32) u16 Pf[4096];   // [4 msub][2 kc][32][16]     8 KB
  __shared__ float Ss[64 * 65];
  __shared__ float mstate[64], lstate[64], alphas[64];
  __shared__ int qdoc[64], kdoc[64];

  const int bh = blockIdx.x, b = bh >> 4, h = bh & 15;
  const int qi = blockIdx.y;
  const int tid = threadIdx.x, wave = tid >> 5, wm = wave & 3, wn = wave >> 2;
  const int lane = tid & 31, kh = lane >> 4, nn = lane & 15;
  const float scale = 0.08838834764831845f; // 1/sqrt(128)
  const unsigned kt_base = lds_addr(Ktf);

  if (tid < 64) {
    qdoc[tid] = doc[b * SEQ_T + 64 * qi + tid];
    mstate[tid] = -1e30f;
    lstate[tid] = 0.0f;
  }

  bh16 qa[4];                                // Q fragments in registers
  const u16* qbase =
      qb + ((size_t)(b * NUM_HEADS + h) * SEQ_T + (64 * qi + 16 * wm)) * HEAD_DIM;
  #pragma unroll
  for (int kc = 0; kc < 4; ++kc) qa[kc] = gather_a_frag(qbase + kc * 32, HEAD_DIM);

  f32x8 oacc[4];
  #pragma unroll
  for (int t = 0; t < 4; ++t) oacc[t] = zero8();
  __syncthreads();

  for (int j = 0; j <= qi; ++j) {
    const int s0 = 64 * j;
    // --- K^T tile: frag-major, async copy (lane blocks contiguous along d).
    #pragma unroll
    for (int it = 0; it < 4; ++it) {
      int d16  = it * 256 + tid;             // 16-byte block id 0..1023
      int tsk  = d16 >> 1, half = d16 & 1;
      int dchunk = tsk >> 7, ns = (tsk >> 5) & 3, lf = tsk & 31;
      int s    = s0 + ns * 16 + (lf & 15);
      int dloc = dchunk * 32 + ((lf >> 4) << 3) + half * 16;
      async_b128(kt_base + d16 * 16,
                 kb + (size_t)(b * SEQ_T + s) * HEAD_DIM + dloc);
    }
    // --- V tile: frag-major, scalar (lane elems strided in global).
    #pragma unroll
    for (int i = 0; i < 32; ++i) {
      int e = tid + 256 * i;                 // 64 x 128
      int s = e >> 7, d = e & 127;
      int kc = s >> 5, kk = s & 31;
      int dst = ((kc * 8 + (d >> 4)) * 32 + (d & 15) + (((kk >> 3) & 1) << 4)) * 16 +
                a_elem(kk);
      Vf[dst] = vb[((size_t)(b * SEQ_T + s0 + s)) * HEAD_DIM + d];
    }
    if (tid < 64) kdoc[tid] = doc[b * SEQ_T + s0 + tid];
    wait_async0();
    __syncthreads();

    // --- S = Q K^T : wave computes 16x32, K-dim 128 (4 chunks).
    f32x8 sacc[2];
    sacc[0] = zero8(); sacc[1] = zero8();
    #pragma unroll
    for (int kc = 0; kc < 4; ++kc) {
      #pragma unroll
      for (int t = 0; t < 2; ++t) {
        bh16 bf = load_frag(Ktf + (kc * 4 + wn * 2 + t) * 512);
        sacc[t] = WMMA_BF16(qa[kc], bf, sacc[t]);
      }
    }
    // scale + causal/doc mask -> Ss (row-major f32)
    #pragma unroll
    for (int t = 0; t < 2; ++t)
      #pragma unroll
      for (int i = 0; i < 8; ++i) {
        int m = i + 8 * kh;
        int qrow = 64 * qi + 16 * wm + m;
        int sc = 32 * wn + 16 * t + nn;
        bool ok = ((s0 + sc) <= qrow) && (kdoc[sc] == qdoc[16 * wm + m]);
        Ss[(16 * wm + m) * 65 + sc] = ok ? sacc[t][i] * scale : -1e30f;
      }
    __syncthreads();

    // --- Online softmax, one thread per query row; P written frag-major.
    if (tid < 64) {
      int r = tid, msub = r >> 4, m = r & 15;
      float mo = mstate[r], lo = lstate[r];
      float rm = -1e30f;
      #pragma unroll 4
      for (int c = 0; c < 64; ++c) rm = fmaxf(rm, Ss[r * 65 + c]);
      float nm = fmaxf(mo, rm);
      float al = __expf(mo - nm);
      float rs = 0.0f;
      #pragma unroll 4
      for (int c = 0; c < 64; ++c) {
        float sv = Ss[r * 65 + c];
        float p = (sv <= -1e29f) ? 0.0f : __expf(sv - nm);
        rs += p;
        int kc = c >> 5, kk = c & 31;
        Pf[((msub * 2 + kc) * 32 + m + (((kk >> 3) & 1) << 4)) * 16 + a_elem(kk)] =
            f2bf(p);
      }
      mstate[r] = nm;
      lstate[r] = al * lo + rs;
      alphas[r] = al;
    }
    __syncthreads();

    // --- O = alpha*O + P V : wave owns 16 rows x 64 d, K-dim 64 (2 chunks).
    float af[8];
    #pragma unroll
    for (int i = 0; i < 8; ++i) af[i] = alphas[16 * wm + i + 8 * kh];
    #pragma unroll
    for (int t = 0; t < 4; ++t)
      #pragma unroll
      for (int i = 0; i < 8; ++i) oacc[t][i] *= af[i];
    #pragma unroll
    for (int kc = 0; kc < 2; ++kc) {
      bh16 pa = load_frag(Pf + (wm * 2 + kc) * 512);
      #pragma unroll
      for (int t = 0; t < 4; ++t) {
        bh16 bf = load_frag(Vf + (kc * 8 + wn * 4 + t) * 512);
        oacc[t] = WMMA_BF16(pa, bf, oacc[t]);
      }
    }
    __syncthreads();
  }

  // Normalize; emit bf16 (B*T, D_MODEL) for the out-projection.
  float lf[8];
  #pragma unroll
  for (int i = 0; i < 8; ++i) lf[i] = lstate[16 * wm + i + 8 * kh];
  #pragma unroll
  for (int t = 0; t < 4; ++t)
    #pragma unroll
    for (int i = 0; i < 8; ++i) {
      int m = i + 8 * kh;
      int trow = 64 * qi + 16 * wm + m;
      ao[((size_t)(b * SEQ_T + trow)) * D_MODEL + h * HEAD_DIM + 64 * wn +
         16 * t + nn] = f2bf(oacc[t][i] / lf[i]);
    }
}

// ---------------------------------------------------------------------------
// Kernel 3: out = attn_out @ W_out (bf16 WMMA, f32 out). grid (64, 16).
// A tiles (already bf16) staged frag-major with async-to-LDS.
// ---------------------------------------------------------------------------
__global__ __launch_bounds__(256) void oproj_kernel(
    const u16* __restrict__ ao, const float* __restrict__ Wout,
    float* __restrict__ out) {
  __shared__ __align__(32) u16 Af[2048];     // [4 sub][32][16]
  __shared__ __align__(32) u16 Bf[4096];     // [8 sub][32][16]

  const int row0 = blockIdx.x * 64;
  const int n0   = blockIdx.y * 128;
  const int tid  = threadIdx.x;
  const int wave = tid >> 5, wm = wave & 3, wn = wave >> 2;
  const int lane = tid & 31, kh = lane >> 4, nn = lane & 15;
  const unsigned af_base = lds_addr(Af);

  f32x8 acc[4];
  #pragma unroll
  for (int t = 0; t < 4; ++t) acc[t] = zero8();

  for (int k0 = 0; k0 < D_MODEL; k0 += 32) {
    {                                        // A: async frag-major copy (4 KB)
      int tsk = tid >> 1, half = tid & 1;
      int sub = tsk >> 5, lf = tsk & 31;
      int row = row0 + sub * 16 + (lf & 15);
      int kidx = k0 + half * 16 + ((lf >> 4) << 3);
      async_b128(af_base + tid * 16, ao + (size_t)row * D_MODEL + kidx);
    }
    #pragma unroll
    for (int i = 0; i < 16; ++i) {           // B: convert f32 -> frag-major
      int e = tid + 256 * i;
      int r = e >> 7, c = e & 127;           // r=k, c=n
      int dst = ((c >> 4) * 32 + (c & 15) + (((r >> 3) & 1) << 4)) * 16 + a_elem(r);
      Bf[dst] = f2bf(Wout[(size_t)(k0 + r) * D_MODEL + n0 + c]);
    }
    wait_async0();
    __syncthreads();
    bh16 a = load_frag(Af + wm * 512);
    #pragma unroll
    for (int t = 0; t < 4; ++t) {
      bh16 b = load_frag(Bf + (wn * 4 + t) * 512);
      acc[t] = WMMA_BF16(a, b, acc[t]);
    }
    __syncthreads();
  }

  #pragma unroll
  for (int t = 0; t < 4; ++t)
    #pragma unroll
    for (int i = 0; i < 8; ++i) {
      int m = i + 8 * kh;
      out[(size_t)(row0 + 16 * wm + m) * D_MODEL + n0 + 64 * wn + 16 * t + nn] =
          acc[t][i];
    }
}

// ---------------------------------------------------------------------------
extern "C" void kernel_launch(void* const* d_in, const int* in_sizes, int n_in,
                              void* d_out, int out_size, void* d_ws,
                              size_t ws_size, hipStream_t stream) {
  const float* x    = (const float*)d_in[0];
  const float* sinp = (const float*)d_in[1];
  const float* cosp = (const float*)d_in[2];
  const int*   doc  = (const int*)d_in[3];
  const float* Wqkv = (const float*)d_in[4];
  const float* Wout = (const float*)d_in[5];
  float* out = (float*)d_out;

  char* ws = (char*)d_ws;
  u16* qb = (u16*)ws;                                   // 16 MB (B,H,T,128)
  u16* kb = (u16*)(ws + (size_t)16 * 1024 * 1024);      //  1 MB (B,T,128)
  u16* vb = (u16*)(ws + (size_t)17 * 1024 * 1024);      //  1 MB (B,T,128)
  u16* ao = (u16*)(ws + (size_t)18 * 1024 * 1024);      // 16 MB (B*T,2048)

  dim3 blk(256);
  qkv_rope_kernel<<<dim3(64, 18), blk, 0, stream>>>(x, Wqkv, sinp, cosp, qb, kb, vb);
  attn_kernel<<<dim3(32, 32), blk, 0, stream>>>(qb, kb, vb, doc, ao);
  oproj_kernel<<<dim3(64, 16), blk, 0, stream>>>(ao, Wout, out);
}
